// SparseDecoder_33500744909536
// MI455X (gfx1250) — matrix operand order repaired and checked by reading, not compiled, not run on hardware
//
#include <hip/hip_runtime.h>
#include <hip/hip_bf16.h>

// ---------------------------------------------------------------------------
// Types for CDNA5 WMMA / TDM
// ---------------------------------------------------------------------------
typedef __attribute__((ext_vector_type(16))) __bf16 v16bf;
typedef __attribute__((ext_vector_type(8)))  float  v8f;
typedef __attribute__((ext_vector_type(4)))  float  f4;
typedef unsigned int u32x4 __attribute__((ext_vector_type(4)));
typedef int          i32x8 __attribute__((ext_vector_type(8)));
typedef int          i32x4 __attribute__((ext_vector_type(4)));

#define D_MODEL 512
#define N_HEADS 8
#define D_HEAD  64
#define D_FF    2048
#define L_Q     22
#define L_K     4096
#define BATCH   32
#define SAMPLE_K 45     // 5*ceil(ln(4096)) = 45
#define N_TOP    20     // min(22, 5*ceil(ln(22))) = 20
#define EPS     1e-5f

#define AS_STRIDE 36    // padded f32 row stride for A tile (bank spread)
#define WT_STRIDE 40    // padded bf16 row stride for W^T tile (bank spread)
#define KSTEPS    (D_MODEL / 32)

#if defined(__HIP_DEVICE_COMPILE__) && __has_builtin(__builtin_amdgcn_tensor_load_to_lds) && __has_builtin(__builtin_amdgcn_s_wait_tensorcnt)
#define USE_TDM 1
#endif

// bf16 <-> f32 helpers: native casts (backend emits hw cvt on gfx1250)
__device__ __forceinline__ __bf16 f2bf(float f) { return (__bf16)f; }
__device__ __forceinline__ float  bf2f(__bf16 b) { return (float)b; }
__device__ __forceinline__ float bflo(unsigned u) {
  union { unsigned x; float f; } v; v.x = u << 16; return v.f;
}
__device__ __forceinline__ float bfhi(unsigned u) {
  union { unsigned x; float f; } v; v.x = u & 0xFFFF0000u; return v.f;
}

// ---------------------------------------------------------------------------
// K1: q = PE(22,512) @ Wq + bq   (tiny)
// ---------------------------------------------------------------------------
__global__ __launch_bounds__(256) void qproj_kernel(
    const float* __restrict__ Wq, const float* __restrict__ bq,
    float* __restrict__ qbuf)
{
  const int l = blockIdx.x;            // 0..21
  const int tid = threadIdx.x;
  __shared__ float pe[D_MODEL];
  {
    int i = tid;                       // pair index 0..255
    float div = __expf(-(float)(2 * i) * (9.210340371976184f / (float)D_MODEL));
    float ang = (float)l * div;
    pe[2 * i]     = sinf(ang);
    pe[2 * i + 1] = cosf(ang);
  }
  __syncthreads();
  for (int c = tid; c < D_MODEL; c += 256) {
    float acc = bq[c];
    for (int j = 0; j < D_MODEL; ++j) acc += pe[j] * Wq[(size_t)j * D_MODEL + c];
    qbuf[(size_t)l * D_MODEL + c] = acc;
  }
}

// ---------------------------------------------------------------------------
// K2: dominant GEMM, bf16 WMMA with double-buffered TDM staging.
//   C[M,512] = bf16(A[M,512]) @ bf16(W[512,512]) + bias  (C stored bf16)
// Block = 256 thr = 8 waves. Tile: M=128, N=64, Kstep=32.
// A tile DMA'd by the Tensor Data Mover into alternating LDS buffers
// (prefetch next tile, s_wait_tensorcnt(1) -> oldest transfer complete,
//  transfers from one wave retire in order). pad_enable gives the padded
// 36-float row stride for bank spread. W tile staged transposed as bf16.
// NOTE: no local arrays of LDS pointers/offsets — their constant-folded
// initializers (addrspacecast of LDS symbols) are unsupported by ld.lld.
// ---------------------------------------------------------------------------
#ifdef USE_TDM
__device__ __forceinline__ void tdm_issue_a_tile(const float* gptr,
                                                 unsigned ldsOff,
                                                 i32x8 g1) {
  unsigned long long ga = (unsigned long long)(size_t)(const void*)gptr;
  u32x4 g0;
  g0[0] = 1u;                                              // count=1
  g0[1] = ldsOff;                                          // lds_addr
  g0[2] = (unsigned)(ga & 0xFFFFFFFFull);                  // global_addr lo
  g0[3] = ((unsigned)(ga >> 32) & 0x01FFFFFFu) | (2u << 30); // hi | type=2
  const i32x4 z4 = {};
#if __clang_major__ >= 23
  __builtin_amdgcn_tensor_load_to_lds(g0, g1, z4, z4, (i32x8){}, 0);
#else
  __builtin_amdgcn_tensor_load_to_lds(g0, g1, z4, z4, 0);
#endif
}
#endif

__global__ __launch_bounds__(256) void proj_gemm_bf16_wmma(
    const float* __restrict__ A, const float* __restrict__ W,
    const float* __restrict__ bias, __bf16* __restrict__ C)
{
  const int tid  = threadIdx.x;
  const int lane = tid & 31;
  const int wave = tid >> 5;
  const int mb = blockIdx.x * 128;           // tile base row
  const int m0 = mb + wave * 16;             // wave's 16-row strip
  const int n0 = blockIdx.y * 64;
  const int hs  = lane >> 4;                 // half-select
  const int l15 = lane & 15;

#ifdef USE_TDM
  __shared__ __align__(16) float  As0[128 * AS_STRIDE];  // 18 KB
  __shared__ __align__(16) float  As1[128 * AS_STRIDE];  // 18 KB
#else
  __shared__ __align__(16) float  As0[128 * AS_STRIDE];
#endif
  __shared__ __align__(16) __bf16 Wt[64 * WT_STRIDE];    // 5 KB (W^T, bf16)

  v8f acc[4] = {};

#ifdef USE_TDM
  // ---- Tensor DMA descriptor group 1 (constant over K loop) --------------
  // data_size=4B(code 2), pad_enable, pad_interval=32 dw (code 4),
  // pad_amount=4 dw (code 3)  ->  LDS row stride 36 floats.
  i32x8 g1;
  g1[0] = (2 << 16) | (1 << 20) | (4 << 22) | (3 << 25);
  g1[1] = (int)((unsigned)(D_MODEL & 0xFFFF) << 16);         // tensor_dim0 lo16
  g1[2] = (int)(((unsigned)D_MODEL >> 16) & 0xFFFFu);        // dim0 hi | dim1 lo
  g1[2] |= (int)(((unsigned)(BATCH * L_K) & 0xFFFFu) << 16);
  g1[3] = (int)((((unsigned)(BATCH * L_K)) >> 16) & 0xFFFFu) | (32 << 16); // dim1 hi | tile_dim0=32
  g1[4] = 128;                                               // tile_dim1=128, tile_dim2=0
  g1[5] = D_MODEL;                                           // tensor_dim0_stride lo32
  g1[6] = 0;
  g1[7] = 0;
  const unsigned ldsA0 =
      (unsigned)(unsigned long long)(__attribute__((address_space(3))) char*)(void*)As0;
  const unsigned ldsA1 =
      (unsigned)(unsigned long long)(__attribute__((address_space(3))) char*)(void*)As1;
  // prologue: start DMA of first A tile
  if (tid < 32) tdm_issue_a_tile(A + (size_t)mb * D_MODEL, ldsA0, g1);
#endif

  for (int i = 0; i < KSTEPS; ++i) {
    const int k0 = i * 32;
    // ---- stage W tile transposed as bf16 (64 cols x 32 k) ---------------
    for (int t = tid; t < 32 * 64; t += 256) {
      int kk = t >> 6, n = t & 63;
      Wt[n * WT_STRIDE + kk] = f2bf(W[(size_t)(k0 + kk) * D_MODEL + n0 + n]);
    }
#ifdef USE_TDM
    const float* Acur = (i & 1) ? As1 : As0;
    if (tid < 32) {
      if (i + 1 < KSTEPS) {
        // prefetch next A tile into the other buffer, then wait for current
        tdm_issue_a_tile(A + (size_t)mb * D_MODEL + (k0 + 32),
                         ((i + 1) & 1) ? ldsA1 : ldsA0, g1);
        __builtin_amdgcn_s_wait_tensorcnt(1);
      } else {
        __builtin_amdgcn_s_wait_tensorcnt(0);
      }
    }
#else
    const float* Acur = As0;
    for (int t = tid; t < 128 * 8; t += 256) {       // 1024 x float4 chunks
      int idx = t << 2;
      int row = idx >> 5, k = idx & 31;
      f4 vv = *(const f4*)(A + (size_t)(mb + row) * D_MODEL + k0 + k);
      *(f4*)&As0[row * AS_STRIDE + k] = vv;
    }
#endif
    __syncthreads();

    // ---- A fragment: K runs {hs*8 .. +7} and {16+hs*8 .. +7} ------------
    const float* arow = &Acur[(wave * 16 + l15) * AS_STRIDE];
    f4 a0 = *(const f4*)(arow + hs * 8);
    f4 a1 = *(const f4*)(arow + hs * 8 + 4);
    f4 a2 = *(const f4*)(arow + 16 + hs * 8);
    f4 a3 = *(const f4*)(arow + 16 + hs * 8 + 4);
    v16bf a;
#pragma unroll
    for (int j = 0; j < 4; ++j) {
      a[j]      = f2bf(a0[j]);
      a[j + 4]  = f2bf(a1[j]);
      a[j + 8]  = f2bf(a2[j]);
      a[j + 12] = f2bf(a3[j]);
    }

    // ---- B fragments: K runs {hs*8 .. +7} and {16+hs*8 .. +7} at Wt[col] --
#pragma unroll
    for (int t = 0; t < 4; ++t) {
      const int col = t * 16 + l15;
      const __bf16* wrow = &Wt[col * WT_STRIDE];
      v16bf b;
#pragma unroll
      for (int j = 0; j < 16; ++j) b[j] = wrow[(j < 8) ? (hs * 8 + j) : (16 + hs * 8 + j - 8)];
      acc[t] = __builtin_amdgcn_wmma_f32_16x16x32_bf16(
          false, a, false, b, (short)0, acc[t], false, false);
    }
    __syncthreads();   // protects Wt and the A buffer reused 2 steps later
  }

  // ---- epilogue: VGPR r -> M = m0 + r + hs*8, N = n0 + 16t + l15 --------
#pragma unroll
  for (int t = 0; t < 4; ++t) {
    const int col = n0 + t * 16 + l15;
    const float bb = bias[col];
#pragma unroll
    for (int r = 0; r < 8; ++r) {
      const int row = m0 + r + hs * 8;
      C[(size_t)row * D_MODEL + col] = f2bf(acc[t][r] + bb);
    }
  }
}

// ---------------------------------------------------------------------------
// K3: sampled scores + M + top-20 query selection, per (b,h)
// ---------------------------------------------------------------------------
__global__ __launch_bounds__(256) void sample_topk_kernel(
    const __bf16* __restrict__ Kb, const float* __restrict__ qbuf,
    int* __restrict__ topidx)
{
  const int bh = blockIdx.x;          // b*8 + h
  const int b = bh >> 3, h = bh & 7;
  const int tid = threadIdx.x;

  __shared__ float ks[SAMPLE_K * D_HEAD];
  __shared__ float ql[L_Q * D_HEAD];
  __shared__ float Mv[L_Q];

  for (int i = tid; i < SAMPLE_K * D_HEAD; i += 256) {
    int s = i / D_HEAD, d = i % D_HEAD;
    int kk = (s * L_K) / SAMPLE_K;
    ks[i] = bf2f(Kb[((size_t)(b * L_K + kk)) * D_MODEL + h * D_HEAD + d]);
  }
  for (int i = tid; i < L_Q * D_HEAD; i += 256) {
    int q = i / D_HEAD, d = i % D_HEAD;
    ql[i] = qbuf[(size_t)q * D_MODEL + h * D_HEAD + d];
  }
  __syncthreads();

  if (tid < L_Q) {
    float mx = -1e30f, sm = 0.0f;
    for (int s = 0; s < SAMPLE_K; ++s) {
      float dot = 0.0f;
      for (int d = 0; d < D_HEAD; ++d) dot += ql[tid * D_HEAD + d] * ks[s * D_HEAD + d];
      mx = fmaxf(mx, dot);
      sm += dot;
    }
    Mv[tid] = mx - sm * (1.0f / (float)SAMPLE_K);
  }
  __syncthreads();

  if (tid == 0) {
    // exclude the 2 smallest (ties: exclude higher index, matching top_k)
    int i1 = 0;
    for (int i = 1; i < L_Q; ++i)
      if (Mv[i] < Mv[i1] || (Mv[i] == Mv[i1] && i > i1)) i1 = i;
    int i2 = (i1 == 0) ? 1 : 0;
    for (int i = 0; i < L_Q; ++i) {
      if (i == i1) continue;
      if (Mv[i] < Mv[i2] || (Mv[i] == Mv[i2] && i > i2)) i2 = i;
    }
    int cnt = 0;
    for (int i = 0; i < L_Q; ++i)
      if (i != i1 && i != i2) topidx[bh * N_TOP + (cnt++)] = i;
  }
}

// ---------------------------------------------------------------------------
// K4: attention per (b,h): v-mean default rows, softmax(QK^T/8)V for top-20
// ---------------------------------------------------------------------------
__global__ __launch_bounds__(256) void attention_kernel(
    const __bf16* __restrict__ Kb, const __bf16* __restrict__ Vb,
    const float* __restrict__ qbuf, const int* __restrict__ topidx,
    float* __restrict__ ctx)
{
  const int bh = blockIdx.x;
  const int b = bh >> 3, h = bh & 7;
  const int tid = threadIdx.x;
  const int d = tid & 63, ch = tid >> 6;   // 4 key-chunks of 1024

  __shared__ float lg[L_K];
  __shared__ float sq[D_HEAD];
  __shared__ float red[256];
  __shared__ float part[4][D_HEAD];
  __shared__ float vmean[D_HEAD];

  const __bf16* Kp = Kb + (size_t)b * L_K * D_MODEL + h * D_HEAD;
  const __bf16* Vp = Vb + (size_t)b * L_K * D_MODEL + h * D_HEAD;

  // phase 0: mean of V over all 4096 keys
  {
    float acc = 0.0f;
    for (int k = ch * 1024; k < (ch + 1) * 1024; ++k)
      acc += bf2f(Vp[(size_t)k * D_MODEL + d]);
    part[ch][d] = acc;
  }
  __syncthreads();
  if (tid < D_HEAD)
    vmean[tid] = (part[0][tid] + part[1][tid] + part[2][tid] + part[3][tid]) *
                 (1.0f / (float)L_K);
  __syncthreads();
  for (int i = tid; i < L_Q * D_HEAD; i += 256)
    ctx[((size_t)bh * L_Q) * D_HEAD + i] = vmean[i & 63];
  __syncthreads();

  const float scale = 0.125f;  // 1/sqrt(64)

  for (int u = 0; u < N_TOP; ++u) {
    const int qi = topidx[bh * N_TOP + u];
    if (tid < D_HEAD) sq[tid] = qbuf[(size_t)qi * D_MODEL + h * D_HEAD + tid];
    __syncthreads();

    // logits (dword-packed bf16 reads: 2 values per u32)
    for (int k = tid; k < L_K; k += 256) {
      const unsigned* kp = (const unsigned*)(Kp + (size_t)k * D_MODEL);
      float s = 0.0f;
#pragma unroll
      for (int j = 0; j < D_HEAD / 2; ++j) {
        unsigned uu = kp[j];
        s += sq[2 * j] * bflo(uu) + sq[2 * j + 1] * bfhi(uu);
      }
      lg[k] = s * scale;
    }
    __syncthreads();

    // max reduce
    float m = -1e30f;
    for (int k = tid; k < L_K; k += 256) m = fmaxf(m, lg[k]);
    red[tid] = m; __syncthreads();
    for (int st = 128; st > 0; st >>= 1) {
      if (tid < st) red[tid] = fmaxf(red[tid], red[tid + st]);
      __syncthreads();
    }
    m = red[0]; __syncthreads();

    // exp + sum
    float ssum = 0.0f;
    for (int k = tid; k < L_K; k += 256) {
      float e = __expf(lg[k] - m);
      lg[k] = e;
      ssum += e;
    }
    red[tid] = ssum; __syncthreads();
    for (int st = 128; st > 0; st >>= 1) {
      if (tid < st) red[tid] += red[tid + st];
      __syncthreads();
    }
    const float S = red[0]; __syncthreads();

    // ctx accumulate (coalesced over d across lanes)
    float acc = 0.0f;
    for (int k = ch * 1024; k < (ch + 1) * 1024; ++k)
      acc += lg[k] * bf2f(Vp[(size_t)k * D_MODEL + d]);
    part[ch][d] = acc;
    __syncthreads();
    if (tid < D_HEAD)
      ctx[((size_t)bh * L_Q + qi) * D_HEAD + tid] =
          (part[0][tid] + part[1][tid] + part[2][tid] + part[3][tid]) / S;
    __syncthreads();
  }
}

// ---------------------------------------------------------------------------
// K5: attn_out = ctx-row @ Wo + bo ; x = LN(2*attn_out)
// ---------------------------------------------------------------------------
__global__ __launch_bounds__(256) void outproj_ln_kernel(
    const float* __restrict__ ctx, const float* __restrict__ Wo,
    const float* __restrict__ bo, const float* __restrict__ g1,
    const float* __restrict__ be1, float* __restrict__ xbuf)
{
  const int row = blockIdx.x;              // b*22 + l
  const int b = row / L_Q, l = row % L_Q;
  const int tid = threadIdx.x;

  __shared__ float rbuf[D_MODEL];
  __shared__ float srow[D_MODEL];
  __shared__ float red[256];

  for (int j = tid; j < D_MODEL; j += 256) {
    int h = j >> 6, dd = j & 63;
    rbuf[j] = ctx[(((size_t)b * N_HEADS + h) * L_Q + l) * D_HEAD + dd];
  }
  __syncthreads();
  for (int c = tid; c < D_MODEL; c += 256) {
    float acc = bo[c];
    for (int j = 0; j < D_MODEL; ++j) acc += rbuf[j] * Wo[(size_t)j * D_MODEL + c];
    srow[c] = 2.0f * acc;                  // attn_out + attn_out
  }
  __syncthreads();

  float s = 0.0f, sq = 0.0f;
  for (int c = tid; c < D_MODEL; c += 256) { float v = srow[c]; s += v; sq += v * v; }
  red[tid] = s; __syncthreads();
  for (int st = 128; st > 0; st >>= 1) { if (tid < st) red[tid] += red[tid + st]; __syncthreads(); }
  const float mu = red[0] / (float)D_MODEL; __syncthreads();
  red[tid] = sq; __syncthreads();
  for (int st = 128; st > 0; st >>= 1) { if (tid < st) red[tid] += red[tid + st]; __syncthreads(); }
  const float var = red[0] / (float)D_MODEL - mu * mu;
  const float rs = rsqrtf(var + EPS);
  for (int c = tid; c < D_MODEL; c += 256)
    xbuf[(size_t)row * D_MODEL + c] = (srow[c] - mu) * rs * g1[c] + be1[c];
}

// ---------------------------------------------------------------------------
// K6: ff = relu(x @ conv1_w^T + b1)
// ---------------------------------------------------------------------------
__global__ __launch_bounds__(256) void ffn1_kernel(
    const float* __restrict__ xbuf, const float* __restrict__ W1,
    const float* __restrict__ b1, float* __restrict__ ff)
{
  const int row = blockIdx.x;
  const int tid = threadIdx.x;
  __shared__ float xr[D_MODEL];
  for (int j = tid; j < D_MODEL; j += 256) xr[j] = xbuf[(size_t)row * D_MODEL + j];
  __syncthreads();
  for (int c = tid; c < D_FF; c += 256) {
    const float* w = W1 + (size_t)c * D_MODEL;
    float acc = b1[c];
    for (int j = 0; j < D_MODEL; ++j) acc += xr[j] * w[j];
    ff[(size_t)row * D_FF + c] = fmaxf(acc, 0.0f);
  }
}

// ---------------------------------------------------------------------------
// K7: ff2 = ff @ conv2_w^T + b2 ; y = LN(x+ff2) ; forecast = y @ Wout + bout
// out layout: [704] forecasts, then [704*512] y
// ---------------------------------------------------------------------------
__global__ __launch_bounds__(256) void ffn2_ln_out_kernel(
    const float* __restrict__ xbuf, const float* __restrict__ ff,
    const float* __restrict__ W2, const float* __restrict__ b2,
    const float* __restrict__ g2, const float* __restrict__ be2,
    const float* __restrict__ Wout, const float* __restrict__ bout,
    float* __restrict__ out)
{
  const int row = blockIdx.x;
  const int tid = threadIdx.x;
  __shared__ float fr[D_FF];
  __shared__ float srow[D_MODEL];
  __shared__ float red[256];

  for (int j = tid; j < D_FF; j += 256) fr[j] = ff[(size_t)row * D_FF + j];
  __syncthreads();
  for (int c = tid; c < D_MODEL; c += 256) {
    const float* w = W2 + (size_t)c * D_FF;
    float acc = b2[c];
    for (int j = 0; j < D_FF; ++j) acc += fr[j] * w[j];
    srow[c] = xbuf[(size_t)row * D_MODEL + c] + acc;
  }
  __syncthreads();

  float s = 0.0f, sq = 0.0f;
  for (int c = tid; c < D_MODEL; c += 256) { float v = srow[c]; s += v; sq += v * v; }
  red[tid] = s; __syncthreads();
  for (int st = 128; st > 0; st >>= 1) { if (tid < st) red[tid] += red[tid + st]; __syncthreads(); }
  const float mu = red[0] / (float)D_MODEL; __syncthreads();
  red[tid] = sq; __syncthreads();
  for (int st = 128; st > 0; st >>= 1) { if (tid < st) red[tid] += red[tid + st]; __syncthreads(); }
  const float var = red[0] / (float)D_MODEL - mu * mu;
  const float rs = rsqrtf(var + EPS);
  __syncthreads();

  float fsum = 0.0f;
  for (int c = tid; c < D_MODEL; c += 256) {
    float y = (srow[c] - mu) * rs * g2[c] + be2[c];
    out[(size_t)(BATCH * L_Q) + (size_t)row * D_MODEL + c] = y;
    fsum += y * Wout[c];
  }
  red[tid] = fsum; __syncthreads();
  for (int st = 128; st > 0; st >>= 1) { if (tid < st) red[tid] += red[tid + st]; __syncthreads(); }
  if (tid == 0) out[row] = red[0] + bout[0];
}

// ---------------------------------------------------------------------------
// Launch
// ---------------------------------------------------------------------------
extern "C" void kernel_launch(void* const* d_in, const int* in_sizes, int n_in,
                              void* d_out, int out_size, void* d_ws, size_t ws_size,
                              hipStream_t stream) {
  const float* enc     = (const float*)d_in[0];
  const float* Wq      = (const float*)d_in[1];
  const float* bq      = (const float*)d_in[2];
  const float* Wk      = (const float*)d_in[3];
  const float* bk      = (const float*)d_in[4];
  const float* Wv      = (const float*)d_in[5];
  const float* bv      = (const float*)d_in[6];
  const float* Wo      = (const float*)d_in[7];
  const float* bo      = (const float*)d_in[8];
  const float* conv1_w = (const float*)d_in[9];
  const float* conv1_b = (const float*)d_in[10];
  const float* conv2_w = (const float*)d_in[11];
  const float* conv2_b = (const float*)d_in[12];
  const float* gamma1  = (const float*)d_in[13];
  const float* beta1   = (const float*)d_in[14];
  const float* gamma2  = (const float*)d_in[15];
  const float* beta2   = (const float*)d_in[16];
  const float* Wout    = (const float*)d_in[17];
  const float* bout    = (const float*)d_in[18];
  float* out = (float*)d_out;

  // workspace carve-up
  const size_t MTOK = (size_t)BATCH * L_K;         // 131072 token rows
  __bf16* Kb   = (__bf16*)d_ws;                    // MTOK*512 bf16
  __bf16* Vb   = Kb + MTOK * D_MODEL;              // MTOK*512 bf16
  float*  qbuf = (float*)(Vb + MTOK * D_MODEL);    // 22*512 f32
  int*    topi = (int*)(qbuf + (size_t)L_Q * D_MODEL);         // 32*8*20 int
  float*  ctx  = (float*)(topi + (size_t)BATCH * N_HEADS * N_TOP);
  float*  xbuf = ctx + (size_t)BATCH * N_HEADS * L_Q * D_HEAD;
  float*  ffb  = xbuf + (size_t)BATCH * L_Q * D_MODEL;

  // 1) q projection (tiny)
  qproj_kernel<<<L_Q, 256, 0, stream>>>(Wq, bq, qbuf);

  // 2) dominant WMMA GEMMs: K and V projections -> bf16
  dim3 ggrid(MTOK / 128, D_MODEL / 64);
  proj_gemm_bf16_wmma<<<ggrid, 256, 0, stream>>>(enc, Wk, bk, Kb);
  proj_gemm_bf16_wmma<<<ggrid, 256, 0, stream>>>(enc, Wv, bv, Vb);

  // 3) sampled scores + top-k query selection
  sample_topk_kernel<<<BATCH * N_HEADS, 256, 0, stream>>>(Kb, qbuf, topi);

  // 4) sparse attention
  attention_kernel<<<BATCH * N_HEADS, 256, 0, stream>>>(Kb, Vb, qbuf, topi, ctx);

  // 5) output projection + LN1
  outproj_ln_kernel<<<BATCH * L_Q, 256, 0, stream>>>(ctx, Wo, bo, gamma1, beta1, xbuf);

  // 6) FFN1 (relu)
  ffn1_kernel<<<BATCH * L_Q, 256, 0, stream>>>(xbuf, conv1_w, conv1_b, ffb);

  // 7) FFN2 + LN2 + forecast head
  ffn2_ln_out_kernel<<<BATCH * L_Q, 256, 0, stream>>>(
      xbuf, ffb, conv2_w, conv2_b, gamma2, beta2, Wout, bout, out);
}
